// RadianceNetwork_56985626083649
// MI455X (gfx1250) — matrix operand order repaired and checked by reading, not compiled, not run on hardware
//
#include <hip/hip_runtime.h>
#include <hip/hip_bf16.h>

typedef __attribute__((ext_vector_type(16))) _Float16 v16h;
typedef __attribute__((ext_vector_type(8)))  _Float16 v8h;
typedef __attribute__((ext_vector_type(8)))  float    v8f;

// ---------------------------------------------------------------------------
// Weight packing: fp32 [N,K] (torch Linear weight) -> f16 WMMA B-fragment
// stream. Layout per 16x16x32 tile (ct = N/16 tile, kc = K/32 chunk):
//   offset = ((ct*KC + kc)*32 + lane)*16 + e
//   lane 0-15 : column n = ct*16+lane,      K = kc*32 + e        (e=0..15)
//   lane16-31 : column n = ct*16+lane-16,   K = kc*32 + 16 + e
// K/N padded with zeros (handles 131->160, 129->144).
// ---------------------------------------------------------------------------
__global__ void pack_weights(const float* __restrict__ W, _Float16* __restrict__ dst,
                             int Kreal, int Ktot, int Nreal, int total) {
    int KC = Ktot >> 5;
    for (int idx = blockIdx.x * blockDim.x + threadIdx.x; idx < total;
         idx += gridDim.x * blockDim.x) {
        int e    = idx & 15;
        int lane = (idx >> 4) & 31;
        int rest = idx >> 9;              // ct*KC + kc
        int kc   = rest % KC;
        int ct   = rest / KC;
        int col  = ct * 16 + (lane & 15);
        int k    = kc * 32 + ((lane < 16) ? 0 : 16) + e;
        float v  = (k < Kreal && col < Nreal) ? W[(size_t)col * Kreal + k] : 0.0f;
        dst[idx] = (_Float16)v;
    }
}

// ---------------------------------------------------------------------------
// One WMMA MLP layer, register-blocked over M: each wave owns 16-column
// strips (ct) and computes all 4 row tiles with 4 f32 accumulators, so each
// B fragment (global/L2) is loaded once and fed to 4 WMMAs. A fragments come
// from LDS (2 x ds_load_b128 each). Optional fused sin(freq*x) activation.
// ---------------------------------------------------------------------------
template <int KTOT, int NREAL, int NTOT, int INSTR, int OUTSTR, bool ACT>
__device__ __forceinline__ void mlp_layer(const _Float16* __restrict__ wpack,
                                          const float* __restrict__ bias,
                                          const float* __restrict__ freq,
                                          const _Float16* actIn,
                                          _Float16* actOut) {
    const int tid  = threadIdx.x;
    const int lane = tid & 31;
    const int wave = tid >> 5;
    constexpr int KC = KTOT / 32;
    constexpr int CT = NTOT / 16;
    const int lrow = lane & 15;
    const int koff = (lane < 16) ? 0 : 16;  // A-frag K split, contiguous k = 16*h + s (as in B pack)

    __builtin_prefetch(wpack + (size_t)lane * 16, 0, 0);

    // A-row base pointers for the 4 row tiles (rows rt*16 + lrow)
    const _Float16* arow[4];
#pragma unroll
    for (int rt = 0; rt < 4; ++rt)
        arow[rt] = actIn + (rt * 16 + lrow) * INSTR + koff;

    for (int ct = wave; ct < CT; ct += 8) {
        v8f acc[4];
#pragma unroll
        for (int rt = 0; rt < 4; ++rt)
            acc[rt] = (v8f){0.f, 0.f, 0.f, 0.f, 0.f, 0.f, 0.f, 0.f};

        const _Float16* bptr = wpack + (size_t)ct * KC * 512 + lane * 16;
#pragma unroll
        for (int kc = 0; kc < KC; ++kc) {
            v16h b = *(const v16h*)(bptr + kc * 512);   // 1 B load ...
#pragma unroll
            for (int rt = 0; rt < 4; ++rt) {            // ... 4 WMMAs
                // A: lane<16 holds K {0..15}, lane>=16 holds {16..31}
                v8h lo = *(const v8h*)(arow[rt] + kc * 32);
                v8h hi = *(const v8h*)(arow[rt] + kc * 32 + 8);
                v16h a = __builtin_shufflevector(lo, hi, 0, 1, 2, 3, 4, 5, 6, 7,
                                                 8, 9, 10, 11, 12, 13, 14, 15);
                acc[rt] = __builtin_amdgcn_wmma_f32_16x16x32_f16(
                    false, a, false, b, (short)0, acc[rt], false, false);
            }
        }

        const int col  = ct * 16 + lrow;
        const int colc = (col < NREAL) ? col : (NREAL - 1);
        const float bv = bias[colc];
        const float fv = ACT ? freq[colc] : 0.0f;
#pragma unroll
        for (int rt = 0; rt < 4; ++rt) {
#pragma unroll
            for (int i = 0; i < 8; ++i) {
                // C/D layout: VGPR i -> row M = i (lanes 0-15) or 8+i (lanes 16-31)
                int row = rt * 16 + ((lane < 16) ? i : (8 + i));
                float v = acc[rt][i] + bv;
                if (ACT) v = __sinf(fv * v);
                actOut[row * OUTSTR + col] = (_Float16)v;
            }
        }
    }
}

// ---------------------------------------------------------------------------
// Fused radiance network. 64 samples per block. LDS overlay (160 KB total):
//   H0   [0      ,131072)  64x1024 f16     (d1 layer0 out)
//   H1   [131072 ,163840)  64x256
//   H2   [0      , 16384)  64x128          (H0 dead)
//   CAT1 [16384  , 36864)  64x160  z1|pos|0  (persists through d2)
//   G0   [40960  ,106496)  64x512
//   G1   [106496 ,139264)  64x256          (H1 dead)
//   G2   [0      , 16384)  64x128          (H2 dead)
//   Z2   [40960  , 59392)  64x144          (G0 dead)
//   CAT2 [61440  , 81920)  64x160  z1+z2|dir|0
//   C0   [81920  ,114688)  64x256
// ---------------------------------------------------------------------------
__global__ __launch_bounds__(256) void radiance_fwd(
    const float* __restrict__ pos, const float* __restrict__ dir,
    const float* __restrict__ d1_W0, const float* __restrict__ d1_b0, const float* __restrict__ d1_f0,
    const _Float16* __restrict__ w1, const float* __restrict__ d1_b1, const float* __restrict__ d1_f1,
    const _Float16* __restrict__ w2, const float* __restrict__ d1_b2, const float* __restrict__ d1_f2,
    const _Float16* __restrict__ w3, const float* __restrict__ d1_b3, const float* __restrict__ d1_f3,
    const _Float16* __restrict__ w4, const float* __restrict__ d2_b0, const float* __restrict__ d2_f0,
    const _Float16* __restrict__ w5, const float* __restrict__ d2_b1, const float* __restrict__ d2_f1,
    const _Float16* __restrict__ w6, const float* __restrict__ d2_b2, const float* __restrict__ d2_f2,
    const _Float16* __restrict__ w7, const float* __restrict__ d2_b3,
    const _Float16* __restrict__ w8, const float* __restrict__ c_b0, const float* __restrict__ c_f0,
    const float* __restrict__ c_W1, const float* __restrict__ c_b1,
    float* __restrict__ out, int Ntotal) {
    extern __shared__ char smem[];
    _Float16* H0   = (_Float16*)(smem);
    _Float16* H1   = (_Float16*)(smem + 131072);
    _Float16* H2   = (_Float16*)(smem);
    _Float16* CAT1 = (_Float16*)(smem + 16384);
    _Float16* G0   = (_Float16*)(smem + 40960);
    _Float16* G1   = (_Float16*)(smem + 106496);
    _Float16* G2   = (_Float16*)(smem);
    _Float16* Z2   = (_Float16*)(smem + 40960);
    _Float16* CAT2 = (_Float16*)(smem + 61440);
    _Float16* C0   = (_Float16*)(smem + 81920);

    const int tid  = threadIdx.x;
    const int row0 = blockIdx.x * 64;

    // ---- d1 layer0: 3 -> 1024, pure VALU (K=3 would waste WMMA) ----
    for (int idx = tid; idx < 64 * 1024; idx += 256) {
        int row = idx >> 10, col = idx & 1023;
        const float* p = pos + (size_t)(row0 + row) * 3;
        const float* w = d1_W0 + (size_t)col * 3;
        float v = p[0] * w[0] + p[1] * w[1] + p[2] * w[2] + d1_b0[col];
        H0[idx] = (_Float16)__sinf(d1_f0[col] * v);
    }
    __syncthreads();

    mlp_layer<1024, 256, 256, 1024, 256, true>(w1, d1_b1, d1_f1, H0, H1);
    __syncthreads();
    mlp_layer<256, 128, 128, 256, 128, true>(w2, d1_b2, d1_f2, H1, H2);
    __syncthreads();

    // fill concat(z1,pos) pad: cols 128..130 = pos, 131..159 = 0 (keep finite!)
    for (int idx = tid; idx < 64 * 32; idx += 256) {
        int row = idx >> 5, c = 128 + (idx & 31);
        float v = (c < 131) ? pos[(size_t)(row0 + row) * 3 + (c - 128)] : 0.0f;
        CAT1[row * 160 + c] = (_Float16)v;
    }
    mlp_layer<128, 128, 128, 128, 160, true>(w3, d1_b3, d1_f3, H2, CAT1); // z1 -> CAT1[:,0:128]
    __syncthreads();

    mlp_layer<160, 512, 512, 160, 512, true>(w4, d2_b0, d2_f0, CAT1, G0);
    __syncthreads();
    mlp_layer<512, 256, 256, 512, 256, true>(w5, d2_b1, d2_f1, G0, G1);
    __syncthreads();
    mlp_layer<256, 128, 128, 256, 128, true>(w6, d2_b2, d2_f2, G1, G2);
    __syncthreads();
    mlp_layer<128, 129, 144, 128, 144, false>(w7, d2_b3, nullptr, G2, Z2); // z2, no act
    __syncthreads();

    // density = softplus(z2[:,128] - 1), threshold 8
    if (tid < 64) {
        float s = (float)Z2[tid * 144 + 128] - 1.0f;
        float d = (s > 8.0f) ? s : log1pf(__expf(fminf(s, 8.0f)));
        out[(size_t)3 * Ntotal + row0 + tid] = d;
    }
    // color input: cols 0..127 = z1+z2, 128..130 = dir, 131..159 = 0
    for (int idx = tid; idx < 64 * 160; idx += 256) {
        int row = idx / 160, c = idx - row * 160;
        float v;
        if (c < 128)      v = (float)CAT1[row * 160 + c] + (float)Z2[row * 144 + c];
        else if (c < 131) v = dir[(size_t)(row0 + row) * 3 + (c - 128)];
        else              v = 0.0f;
        CAT2[row * 160 + c] = (_Float16)v;
    }
    __syncthreads();

    mlp_layer<160, 256, 256, 160, 256, true>(w8, c_b0, c_f0, CAT2, C0);
    __syncthreads();

    // rgb head: 256 -> 3, sigmoid (192 dot products per block, VALU)
    if (tid < 192) {
        int row = tid / 3, ch = tid - row * 3;
        const float* wr = c_W1 + (size_t)ch * 256;
        float acc = c_b1[ch];
        for (int k = 0; k < 256; ++k)
            acc += (float)C0[row * 256 + k] * wr[k];
        out[(size_t)(row0 + row) * 3 + ch] = 1.0f / (1.0f + __expf(-acc));
    }
}

// ---------------------------------------------------------------------------
extern "C" void kernel_launch(void* const* d_in, const int* in_sizes, int n_in,
                              void* d_out, int out_size, void* d_ws, size_t ws_size,
                              hipStream_t stream) {
    // input order: pos, dir, then params in dict order
    const float* pos   = (const float*)d_in[0];
    const float* dir   = (const float*)d_in[1];
    const float* d1_W0 = (const float*)d_in[2];
    const float* d1_b0 = (const float*)d_in[3];
    const float* d1_f0 = (const float*)d_in[4];
    const float* d1_b1 = (const float*)d_in[6];
    const float* d1_f1 = (const float*)d_in[7];
    const float* d1_b2 = (const float*)d_in[9];
    const float* d1_f2 = (const float*)d_in[10];
    const float* d1_b3 = (const float*)d_in[12];
    const float* d1_f3 = (const float*)d_in[13];
    const float* d2_b0 = (const float*)d_in[15];
    const float* d2_f0 = (const float*)d_in[16];
    const float* d2_b1 = (const float*)d_in[18];
    const float* d2_f1 = (const float*)d_in[19];
    const float* d2_b2 = (const float*)d_in[21];
    const float* d2_f2 = (const float*)d_in[22];
    const float* d2_b3 = (const float*)d_in[24];
    const float* c_b0  = (const float*)d_in[26];
    const float* c_f0  = (const float*)d_in[27];
    const float* c_W1  = (const float*)d_in[28];
    const float* c_b1  = (const float*)d_in[29];

    // GEMM weights to pack (indices into d_in)
    const float* Wsrc[8] = {(const float*)d_in[5],  (const float*)d_in[8],
                            (const float*)d_in[11], (const float*)d_in[14],
                            (const float*)d_in[17], (const float*)d_in[20],
                            (const float*)d_in[23], (const float*)d_in[25]};
    const int Kreal[8] = {1024, 256, 128, 131, 512, 256, 128, 131};
    const int Ktot[8]  = {1024, 256, 128, 160, 512, 256, 128, 160};
    const int Nreal[8] = {256, 128, 128, 512, 256, 128, 129, 256};
    const int Ntot[8]  = {256, 128, 128, 512, 256, 128, 144, 256};

    _Float16* ws = (_Float16*)d_ws;
    _Float16* wp[8];
    size_t off = 0;
    for (int i = 0; i < 8; ++i) {
        wp[i] = ws + off;
        size_t sz = (size_t)(Ntot[i] / 16) * (Ktot[i] / 32) * 512;
        int total = (int)sz;
        int grid  = (total + 255) / 256;
        pack_weights<<<grid, 256, 0, stream>>>(Wsrc[i], wp[i], Kreal[i], Ktot[i],
                                               Nreal[i], total);
        off += sz;
    }

    const int N = in_sizes[0] / 3;          // 262144
    const int blocks = (N + 63) / 64;       // 64 samples per block
    const size_t smem_bytes = 163840;       // 160 KB of the 320 KB WGP LDS

    radiance_fwd<<<blocks, 256, smem_bytes, stream>>>(
        pos, dir,
        d1_W0, d1_b0, d1_f0,
        wp[0], d1_b1, d1_f1,
        wp[1], d1_b2, d1_f2,
        wp[2], d1_b3, d1_f3,
        wp[3], d2_b0, d2_f0,
        wp[4], d2_b1, d2_f1,
        wp[5], d2_b2, d2_f2,
        wp[6], d2_b3,
        wp[7], c_b0, c_f0,
        c_W1, c_b1,
        (float*)d_out, N);
}
